// MHA_78443282694426
// MI455X (gfx1250) — matrix operand order repaired and checked by reading, not compiled, not run on hardware
//
#include <hip/hip_runtime.h>
#include <string.h>

// ---------------- problem constants ----------------
#define HIDDEN   1024
#define HEADS    16
#define HEAD_DIM 64
#define BATCH    4
#define SEQ      4096
#define M_TOTAL  (BATCH * SEQ)      // 16384 rows for the projections
#define EPSF     0.001f
#define CHUNKS   32                 // split-K chunks over S for kv aggregation
#define SCHUNK   (SEQ / CHUNKS)     // 128 rows of S per chunk

// ---------------- CDNA5 WMMA types ----------------
typedef __attribute__((ext_vector_type(16))) __bf16 v16bf;
typedef __attribute__((ext_vector_type(8)))  float  v8f;

union Frag16 {              // 16 bf16 = 32 bytes = 8 dwords
    v16bf v;
    unsigned int u[8];
};

// native bf16 conversion (RNE) -> lets the backend use its cvt path
__device__ __forceinline__ unsigned short f2bf(float f) {
    __bf16 h = (__bf16)f;
    return __builtin_bit_cast(unsigned short, h);
}
__device__ __forceinline__ unsigned int pack_bf16(float a, float b) {
    return (unsigned int)f2bf(a) | ((unsigned int)f2bf(b) << 16);
}
__device__ __forceinline__ float bf2f(unsigned short h) {
    return __uint_as_float(((unsigned int)h) << 16);
}

// =====================================================================
// Kernel 1: projection GEMM  P[m,n] = X[m,k] * W[k,n]   (M=16384,K=N=1024)
// bf16 WMMA, fp32 accum; double-buffered LDS (stage i+1 overlaps WMMA i);
// optional relu+eps epilogue; bf16 output to ws.
// Block tile 128x128, K-slice 32, 8 waves (4x2), each wave 32x64.
// =====================================================================
#define LDA 40   // ushort stride (32 + 8 pad), 80B rows -> 16B aligned

__device__ __forceinline__ void stage_tiles(
    const float* __restrict__ X, const float* __restrict__ W,
    int m0, int n0, int k0, int tid,
    unsigned short (*__restrict__ As)[LDA],
    unsigned short (*__restrict__ Bt)[LDA])
{
    // ---- A tile: 128 rows x 32 k, fp32 -> packed bf16 (ds_store_b32) ----
#pragma unroll
    for (int i = 0; i < 4; ++i) {
        int idx = tid + i * 256;            // 0..1023 float4s
        int r   = idx >> 3;                 // row 0..127
        int c4  = (idx & 7) << 2;           // col 0..28
        float4 f = *(const float4*)(X + (size_t)(m0 + r) * HIDDEN + k0 + c4);
        *(unsigned int*)&As[r][c4]     = pack_bf16(f.x, f.y);
        *(unsigned int*)&As[r][c4 + 2] = pack_bf16(f.z, f.w);
    }
    // ---- B tile transposed: read pairs of k-rows, pack along k ----
#pragma unroll
    for (int i = 0; i < 2; ++i) {
        int idx = tid + i * 256;            // 0..511 : 16 k-pairs x 32 col-quads
        int r2  = (idx >> 5) << 1;          // even k row 0..30
        int c4  = (idx & 31) << 2;          // n col 0..124
        float4 f0 = *(const float4*)(W + (size_t)(k0 + r2)     * HIDDEN + n0 + c4);
        float4 f1 = *(const float4*)(W + (size_t)(k0 + r2 + 1) * HIDDEN + n0 + c4);
        *(unsigned int*)&Bt[c4 + 0][r2] = pack_bf16(f0.x, f1.x);
        *(unsigned int*)&Bt[c4 + 1][r2] = pack_bf16(f0.y, f1.y);
        *(unsigned int*)&Bt[c4 + 2][r2] = pack_bf16(f0.z, f1.z);
        *(unsigned int*)&Bt[c4 + 3][r2] = pack_bf16(f0.w, f1.w);
    }
    // prefetch next-next A slice into cache (global_prefetch)
    if (k0 + 32 < HIDDEN)
        __builtin_prefetch(X + (size_t)(m0 + (tid >> 1)) * HIDDEN + k0 + 32, 0, 1);
}

__global__ __launch_bounds__(256) void proj_gemm_kernel(
    const float* __restrict__ X, const float* __restrict__ W,
    unsigned short* __restrict__ P, int doRelu)
{
    __shared__ unsigned short As[2][128][LDA];   // A[m][k] bf16, double buffered
    __shared__ unsigned short Bt[2][128][LDA];   // B transposed: Bt[n][k] bf16

    const int tid  = threadIdx.x;
    const int lane = tid & 31;
    const int wave = tid >> 5;
    const int wm   = wave >> 1;      // 0..3
    const int wn   = wave & 1;       // 0..1
    const int l15  = lane & 15;
    const int lhi  = lane >> 4;      // 0 or 1
    const int m0   = blockIdx.y * 128;
    const int n0   = blockIdx.x * 128;
    const int ak   = lhi * 8;        // A lane K base (0 / 8)
    const int bk   = lhi * 8;        // B lane K base (0 / 8)

    v8f acc[2][4];
#pragma unroll
    for (int i = 0; i < 2; ++i)
#pragma unroll
        for (int j = 0; j < 4; ++j) acc[i][j] = {};

    stage_tiles(X, W, m0, n0, 0, tid, As[0], Bt[0]);
    __syncthreads();

    for (int it = 0; it < HIDDEN / 32; ++it) {
        const int cur = it & 1;
        // stage next K-slice into the other buffer (overlaps with WMMA below)
        if (it + 1 < HIDDEN / 32)
            stage_tiles(X, W, m0, n0, (it + 1) * 32, tid, As[cur ^ 1], Bt[cur ^ 1]);

        // ---- fragments + WMMA from current buffer ----
        Frag16 a[2], b[4];
#pragma unroll
        for (int mt = 0; mt < 2; ++mt) {
            const unsigned short* rp = &As[cur][wm * 32 + mt * 16 + l15][0];
            *(uint4*)&a[mt].u[0] = *(const uint4*)(rp + ak);        // K ak..ak+7
            *(uint4*)&a[mt].u[4] = *(const uint4*)(rp + ak + 16);   // K ak+16..ak+23
        }
#pragma unroll
        for (int nt = 0; nt < 4; ++nt) {
            const unsigned short* rp = &Bt[cur][wn * 64 + nt * 16 + l15][0];
            *(uint4*)&b[nt].u[0] = *(const uint4*)(rp + bk);        // K bk..bk+7
            *(uint4*)&b[nt].u[4] = *(const uint4*)(rp + bk + 16);   // K bk+16..bk+23
        }
#pragma unroll
        for (int mt = 0; mt < 2; ++mt)
#pragma unroll
            for (int nt = 0; nt < 4; ++nt)
                acc[mt][nt] = __builtin_amdgcn_wmma_f32_16x16x32_bf16(
                    false, a[mt].v, false, b[nt].v, (short)0, acc[mt][nt], false, false);
        __syncthreads();   // one barrier per K-slice
    }

    // ---- epilogue: optional relu+eps, bf16 store ----
#pragma unroll
    for (int mt = 0; mt < 2; ++mt)
#pragma unroll
        for (int nt = 0; nt < 4; ++nt) {
            int gn = n0 + wn * 64 + nt * 16 + l15;
#pragma unroll
            for (int j = 0; j < 8; ++j) {
                int gm = m0 + wm * 32 + mt * 16 + j + lhi * 8;
                float val = acc[mt][nt][j];
                if (doRelu) val = fmaxf(val, 0.0f) + EPSF;
                P[(size_t)gm * HIDDEN + gn] = f2bf(val);
            }
        }
}

// =====================================================================
// Kernel 2a: partial kvs[m,d] = sum_{s in chunk} K[s,m] * V[s,d]  (WMMA)
// and partial ks[m] = sum_s K[s,m].   One WG per (b*head, chunk).
// =====================================================================
#define LDT 136   // SCHUNK(128) + 8 pad, 272B rows -> 16B aligned

__global__ __launch_bounds__(128) void kv_partial_kernel(
    const unsigned short* __restrict__ Kp, const unsigned short* __restrict__ Vp,
    float* __restrict__ kv_part, float* __restrict__ ks_part)
{
    __shared__ unsigned short Kt[64][LDT];   // Kt[m][s]
    __shared__ unsigned short Vt[64][LDT];   // Vt[d][s]

    const int bn = blockIdx.x;            // 0..63 : b*16+n
    const int ch = blockIdx.y;            // 0..CHUNKS-1
    const int b  = bn >> 4, n = bn & 15;
    const int s0 = ch * SCHUNK;
    const int tid  = threadIdx.x;         // 128 threads, 4 waves
    const int lane = tid & 31;
    const int w    = tid >> 5;
    const int l15  = lane & 15;
    const int lhi  = lane >> 4;

    // stage one S-row per thread, transposed into LDS
    {
        const unsigned short* krow =
            Kp + ((size_t)(b * SEQ + s0 + tid)) * HIDDEN + n * HEAD_DIM;
        const unsigned short* vrow =
            Vp + ((size_t)(b * SEQ + s0 + tid)) * HIDDEN + n * HEAD_DIM;
#pragma unroll
        for (int i = 0; i < 8; ++i) {
            uint4 qk = *(const uint4*)(krow + i * 8);
            uint4 qv = *(const uint4*)(vrow + i * 8);
            unsigned int wk[4] = {qk.x, qk.y, qk.z, qk.w};
            unsigned int wv[4] = {qv.x, qv.y, qv.z, qv.w};
#pragma unroll
            for (int j = 0; j < 4; ++j) {
                int m = i * 8 + j * 2;
                Kt[m + 0][tid] = (unsigned short)(wk[j] & 0xFFFFu);
                Kt[m + 1][tid] = (unsigned short)(wk[j] >> 16);
                Vt[m + 0][tid] = (unsigned short)(wv[j] & 0xFFFFu);
                Vt[m + 1][tid] = (unsigned short)(wv[j] >> 16);
            }
        }
    }
    __syncthreads();

    // partial ks sum (threads 0..63 handle one m each)
    if (tid < 64) {
        float s = 0.0f;
        for (int j = 0; j < SCHUNK; ++j) s += bf2f(Kt[tid][j]);
        ks_part[(size_t)(bn * CHUNKS + ch) * 64 + tid] = s;
    }

    // WMMA: A[m][s] = Kt, B[s][d] = V (read via Vt[d][s]); reduce over s
    v8f acc[4];
#pragma unroll
    for (int i = 0; i < 4; ++i) acc[i] = {};
    const int ak = lhi * 8;
    const int bk = lhi * 8;
    for (int so = 0; so < SCHUNK; so += 32) {
        Frag16 a;
        const unsigned short* rp = &Kt[w * 16 + l15][0];
        *(uint4*)&a.u[0] = *(const uint4*)(rp + so + ak);
        *(uint4*)&a.u[4] = *(const uint4*)(rp + so + ak + 16);
#pragma unroll
        for (int dt = 0; dt < 4; ++dt) {
            Frag16 bfr;
            const unsigned short* vp = &Vt[dt * 16 + l15][0];
            *(uint4*)&bfr.u[0] = *(const uint4*)(vp + so + bk);
            *(uint4*)&bfr.u[4] = *(const uint4*)(vp + so + bk + 16);
            acc[dt] = __builtin_amdgcn_wmma_f32_16x16x32_bf16(
                false, a.v, false, bfr.v, (short)0, acc[dt], false, false);
        }
    }

    float* outp = kv_part + (size_t)(bn * CHUNKS + ch) * 4096;
#pragma unroll
    for (int dt = 0; dt < 4; ++dt)
#pragma unroll
        for (int j = 0; j < 8; ++j) {
            int m = w * 16 + j + lhi * 8;
            int d = dt * 16 + l15;
            outp[m * 64 + d] = acc[dt][j];
        }
}

// =====================================================================
// Kernel 2b: deterministic reduction of split-K partials
// =====================================================================
__global__ __launch_bounds__(256) void kv_reduce_kernel(
    const float* __restrict__ kv_part, const float* __restrict__ ks_part,
    float* __restrict__ kvs, float* __restrict__ ks_sum)
{
    const int bn = blockIdx.x;
    const int tid = threadIdx.x;
    for (int e = tid; e < 4096; e += 256) {
        float s = 0.0f;
        for (int c = 0; c < CHUNKS; ++c)
            s += kv_part[(size_t)(bn * CHUNKS + c) * 4096 + e];
        kvs[(size_t)bn * 4096 + e] = s;
    }
    if (tid < 64) {
        float s = 0.0f;
        for (int c = 0; c < CHUNKS; ++c)
            s += ks_part[(size_t)(bn * CHUNKS + c) * 64 + tid];
        ks_sum[bn * 64 + tid] = s;
    }
}

// =====================================================================
// Kernel 3: a_v[s,d] = (Q[s,:] . kvs[:,d]) / (Q[s,:] . ks_sum)   (WMMA)
// One WG per (b*head, 128-row S tile). 8 waves, each 16 rows x 64 d.
// =====================================================================
#define LDQ 72   // 64 + 8 pad ushorts; 144B rows -> 16B aligned

__global__ __launch_bounds__(256) void out_kernel(
    const unsigned short* __restrict__ Qp, const float* __restrict__ kvs,
    const float* __restrict__ ks_sum, float* __restrict__ out)
{
    __shared__ unsigned short Qs[128][LDQ];      // Q[s_local][m] bf16
    __shared__ unsigned short Bkv[64][LDQ];      // kvs transposed: Bkv[d][m] bf16
    __shared__ float ksl[64];
    __shared__ float norm[128];

    const int bn = blockIdx.x;            // b*16+n
    const int b  = bn >> 4, n = bn & 15;
    const int s0 = blockIdx.y * 128;
    const int tid  = threadIdx.x;         // 256 threads, 8 waves
    const int lane = tid & 31;
    const int w    = tid >> 5;
    const int l15  = lane & 15;
    const int lhi  = lane >> 4;

    // ---- stage Q block (already bf16): 128 rows x 64 m ----
#pragma unroll
    for (int i = 0; i < 4; ++i) {
        int idx = tid + i * 256;          // 0..1023 uint4s (8 ushorts each)
        int r   = idx >> 3;               // row 0..127
        int c8  = (idx & 7) << 3;         // m 0..56
        *(uint4*)&Qs[r][c8] =
            *(const uint4*)(Qp + ((size_t)(b * SEQ + s0 + r)) * HIDDEN + n * HEAD_DIM + c8);
    }
    // ---- stage kvs transposed (fp32 -> bf16) ----
#pragma unroll
    for (int i = 0; i < 4; ++i) {
        int idx = tid + i * 256;          // 0..1023 float4s
        int m   = idx >> 4;               // 0..63
        int d4  = (idx & 15) << 2;        // 0..60
        float4 f = *(const float4*)(kvs + (size_t)bn * 4096 + m * 64 + d4);
        Bkv[d4 + 0][m] = f2bf(f.x); Bkv[d4 + 1][m] = f2bf(f.y);
        Bkv[d4 + 2][m] = f2bf(f.z); Bkv[d4 + 3][m] = f2bf(f.w);
    }
    if (tid < 64) ksl[tid] = ks_sum[bn * 64 + tid];
    __syncthreads();

    // ---- per-row normalizer ----
    if (tid < 128) {
        float s = 0.0f;
#pragma unroll
        for (int m = 0; m < 64; ++m) s += bf2f(Qs[tid][m]) * ksl[m];
        norm[tid] = s;
    }
    __syncthreads();

    // ---- WMMA: A[s][m] = Qs, B[m][d] via Bkv[d][m]; K = 64 (2 slices) ----
    v8f acc[4];
#pragma unroll
    for (int i = 0; i < 4; ++i) acc[i] = {};
    const int ak = lhi * 8;
    const int bk = lhi * 8;
#pragma unroll
    for (int k0 = 0; k0 < 64; k0 += 32) {
        Frag16 a;
        const unsigned short* rp = &Qs[w * 16 + l15][0];
        *(uint4*)&a.u[0] = *(const uint4*)(rp + k0 + ak);
        *(uint4*)&a.u[4] = *(const uint4*)(rp + k0 + ak + 16);
#pragma unroll
        for (int dt = 0; dt < 4; ++dt) {
            Frag16 bfr;
            const unsigned short* bp = &Bkv[dt * 16 + l15][0];
            *(uint4*)&bfr.u[0] = *(const uint4*)(bp + k0 + bk);
            *(uint4*)&bfr.u[4] = *(const uint4*)(bp + k0 + bk + 16);
            acc[dt] = __builtin_amdgcn_wmma_f32_16x16x32_bf16(
                false, a.v, false, bfr.v, (short)0, acc[dt], false, false);
        }
    }

    // ---- divide by normalizer, store fp32 ----
#pragma unroll
    for (int dt = 0; dt < 4; ++dt)
#pragma unroll
        for (int j = 0; j < 8; ++j) {
            int sl = w * 16 + j + lhi * 8;
            int d  = dt * 16 + l15;
            float val = acc[dt][j] / norm[sl];
            out[((size_t)(b * SEQ + s0 + sl)) * HIDDEN + n * HEAD_DIM + d] = val;
        }
}

// =====================================================================
// launcher
// =====================================================================
extern "C" void kernel_launch(void* const* d_in, const int* in_sizes, int n_in,
                              void* d_out, int out_size, void* d_ws, size_t ws_size,
                              hipStream_t stream) {
    (void)in_sizes; (void)n_in; (void)out_size; (void)ws_size;
    const float* q  = (const float*)d_in[0];
    const float* k  = (const float*)d_in[1];
    const float* v  = (const float*)d_in[2];
    const float* wq = (const float*)d_in[3];
    const float* wk = (const float*)d_in[4];
    const float* wv = (const float*)d_in[5];
    float* out = (float*)d_out;

    // workspace carve-up
    const size_t projElems = (size_t)M_TOTAL * HIDDEN;          // 16,777,216
    unsigned short* Qp = (unsigned short*)d_ws;
    unsigned short* Kp = Qp + projElems;
    unsigned short* Vp = Kp + projElems;
    float* kv_part = (float*)(Vp + projElems);                  // 64*CH*4096
    float* ks_part = kv_part + (size_t)64 * CHUNKS * 4096;      // 64*CH*64
    float* kvs     = ks_part + (size_t)64 * CHUNKS * 64;        // 64*4096
    float* ks_sum  = kvs + (size_t)64 * 4096;                   // 64*64

    dim3 gProj(HIDDEN / 128, M_TOTAL / 128);
    proj_gemm_kernel<<<gProj, 256, 0, stream>>>(q, wq, Qp, 1);
    proj_gemm_kernel<<<gProj, 256, 0, stream>>>(k, wk, Kp, 1);
    proj_gemm_kernel<<<gProj, 256, 0, stream>>>(v, wv, Vp, 0);

    kv_partial_kernel<<<dim3(BATCH * HEADS, CHUNKS), 128, 0, stream>>>(
        Kp, Vp, kv_part, ks_part);
    kv_reduce_kernel<<<BATCH * HEADS, 256, 0, stream>>>(
        kv_part, ks_part, kvs, ks_sum);

    out_kernel<<<dim3(BATCH * HEADS, SEQ / 128), 256, 0, stream>>>(
        Qp, kvs, ks_sum, out);
}
